// LorentzConv2d_kernels_old_21199958573224
// MI455X (gfx1250) — matrix operand order repaired and checked by reading, not compiled, not run on hardware
//
#include <hip/hip_runtime.h>

typedef __attribute__((ext_vector_type(16))) _Float16 v16h;
typedef __attribute__((ext_vector_type(8)))  float    v8f;
typedef int v4i __attribute__((vector_size(16)));

#if __has_builtin(__builtin_amdgcn_global_load_async_to_lds_b128) && \
    __has_builtin(__builtin_amdgcn_s_wait_asynccnt)
#define USE_ASYNC_LDS 1
#endif

#define BB   8
#define HH   56
#define WW   56
#define HP   58
#define WP   58
#define NPIX (BB*HH*WW)            /* 25088 */
#define XPAD_ELEMS (BB*HP*WP*64)   /* 1722368 f16 elems per padded map */

__device__ __forceinline__ void pix_decompose(int p, int& b, int& h, int& w) {
    b = p / (HH*WW);
    int r = p - b*(HH*WW);
    h = r / WW;
    w = r - h*WW;
}

__device__ __forceinline__ int pad_base(int b, int h, int w) {
    // interior pixel (h,w) -> padded (h+1, w+1), 64 channels
    return ((b*HP + h + 1)*WP + (w + 1)) * 64;
}

__global__ void zero_ws_kernel(uint4* __restrict__ ws, int n16) {
    uint4 z; z.x = z.y = z.z = z.w = 0u;
    for (int i = blockIdx.x*blockDim.x + threadIdx.x; i < n16;
         i += gridDim.x*blockDim.x)
        ws[i] = z;
}

__global__ void cvt_kern_kernel(const float* __restrict__ kern,
                                _Float16* __restrict__ kern16) {
    int i = blockIdx.x*blockDim.x + threadIdx.x;   // 4096 elements exactly
    kern16[i] = (_Float16)kern[i];
}

// ---------------------------------------------------------------------------
// Stage 1: dists[p][o] = arccosh(clamp(-<x_p, k_o>_L, 1+eps))^2
// GEMM: M=16 pixels per block, N=64 (4 waves x 16), K=64 (2 WMMA k-steps).
// Channel-0 of x negated in LDS so acc == Minkowski inner directly.
// Also emits f16 copy of x into padded map for stage 2.
// ---------------------------------------------------------------------------
__global__ __launch_bounds__(128)
void lorentz_dists_kernel(const float* __restrict__ x,
                          const _Float16* __restrict__ kern16,
                          _Float16* __restrict__ xpad,
                          _Float16* __restrict__ dpad) {
    __shared__ __align__(16) _Float16 ldsA[16*64];   // x tile, ch0 negated
    __shared__ __align__(16) _Float16 ldsK[64*64];   // kernels f16

    const int tid  = threadIdx.x;
    const int lane = tid & 31;
    const int wave = tid >> 5;
    const int p0   = blockIdx.x * 16;

    __builtin_prefetch(kern16, 0, 3);                // global_prefetch_b8

    for (int i = tid; i < 2048; i += 128)            // 64x64 f16 as u32 pairs
        ((unsigned int*)ldsK)[i] = ((const unsigned int*)kern16)[i];

    for (int i = tid; i < 16*64; i += 128) {
        int pl = i >> 6, c = i & 63;
        int p = p0 + pl;
        if (p < NPIX) {
            int b, h, w; pix_decompose(p, b, h, w);
            float v = x[p*64 + c];
            _Float16 hv = (_Float16)v;
            xpad[pad_base(b, h, w) + c] = hv;
            ldsA[i] = (c == 0) ? (_Float16)(-v) : hv;
        } else {
            ldsA[i] = (_Float16)0.f;
        }
    }
    __syncthreads();

    const int g    = lane >> 4;                 // lane half-group
    const int Mrow = lane & 15;                 // pixel row (A)
    const int ocol = wave*16 + (lane & 15);     // out channel (B/C column)

    v8f acc = {};
    #pragma unroll
    for (int s = 0; s < 2; ++s) {
        const int kbase = s * 32;
        v16h a, bf;
        #pragma unroll
        for (int hh = 0; hh < 16; ++hh) {
            int Ka = ((hh >> 3) << 4) + (g << 3) + (hh & 7);  // A 16-bit layout
            a[hh]  = ldsA[Mrow*64 + kbase + Ka];
            int Kb = ((hh >> 3) << 4) + (g << 3) + (hh & 7);  // B 16-bit layout
            bf[hh] = ldsK[ocol*64 + kbase + Kb];              // B[K=c][N=o]
        }
        acc = __builtin_amdgcn_wmma_f32_16x16x32_f16(
            false, a, false, bf, (short)0, acc, false, false);
    }

    #pragma unroll
    for (int j = 0; j < 8; ++j) {
        int pl = j + (g << 3);                 // C layout: M = j + 8*lanegrp
        int p  = p0 + pl;
        if (p < NPIX) {
            int b, h, w; pix_decompose(p, b, h, w);
            float s = acc[j];                              // Minkowski inner
            float u = fmaxf(-s, 1.0f + 1e-7f);
            float t = __logf(u + __builtin_amdgcn_sqrtf(u*u - 1.0f)); // arccosh
            dpad[pad_base(b, h, w) + ocol] = (_Float16)(t * t);
        }
    }
}

// ---------------------------------------------------------------------------
// Stage 2: per pixel, avgT[c,o] = sum_{k=0..8} x[k,c] * dist[k,o] (K=9 -> 32)
// A = patches (M=c), B = weights (N=o): accumulator rows are c, so the per-o
// reductions over c are mostly in-register (8 adds) + one shfl_xor(16) + one
// LDS atomic per (tile,o). Then Minkowski norm, mean over c>=1, projx over o.
// ---------------------------------------------------------------------------
__global__ __launch_bounds__(128)
void lorentz_agg_kernel(const _Float16* __restrict__ xpad,
                        const _Float16* __restrict__ dpad,
                        float* __restrict__ out) {
    __shared__ __align__(16) _Float16 ldsP[9*64];   // patches x[k][c]
    __shared__ __align__(16) _Float16 ldsW[9*64];   // weights d[k][o]
    __shared__ float s1[64], s2[64], mvec[64];
    __shared__ float msum;

    const int tid  = threadIdx.x;
    const int lane = tid & 31;
    const int wave = tid >> 5;
    const int p    = blockIdx.x;

    int b, h, w; pix_decompose(p, b, h, w);

#ifdef USE_ASYNC_LDS
    // 9 taps x 128B per map, staged via async LDS-direct loads (ASYNCcnt path)
    for (int i = tid; i < 144; i += 128) {
        int arr = (i >= 72);
        int r   = arr ? i - 72 : i;
        int k = r >> 3, j = r & 7;              // tap, 16B chunk
        int kh = k / 3, kw = k - kh*3;
        int off = ((b*HP + h + kh)*WP + (w + kw))*64 + j*8;
        const _Float16* gsrc = (arr ? dpad : xpad) + off;
        _Float16*       ldst = (arr ? ldsW : ldsP) + (k*64 + j*8);
        __builtin_amdgcn_global_load_async_to_lds_b128(
            (v4i*)(void*)gsrc, (v4i*)(void*)ldst, 0, 0);
    }
    __builtin_amdgcn_s_wait_asynccnt(0);
#else
    for (int i = tid; i < 9*32; i += 128) {
        int k = i >> 5, j = i & 31;
        int kh = k / 3, kw = k - kh*3;
        int base = ((b*HP + h + kh)*WP + (w + kw)) * 64;
        ((unsigned int*)ldsP)[i] = ((const unsigned int*)xpad)[(base >> 1) + j];
        ((unsigned int*)ldsW)[i] = ((const unsigned int*)dpad)[(base >> 1) + j];
    }
#endif
    if (tid < 64) { s1[tid] = 0.f; s2[tid] = 0.f; }
    __syncthreads();

    const int g  = lane >> 4;
    const int Mc = wave*16 + (lane & 15);       // A row: in-channel c

    v16h a;                                     // A[c][k] = x[k][c]
    #pragma unroll
    for (int hh = 0; hh < 16; ++hh) {
        int K = ((hh >> 3) << 4) + (g << 3) + (hh & 7);     // A 16-bit layout
        a[hh] = (K < 9) ? ldsP[K*64 + Mc] : (_Float16)0.f;
    }

    #pragma unroll
    for (int t = 0; t < 4; ++t) {
        v16h bf;                                // B[k][o], o-tile t
        #pragma unroll
        for (int hh = 0; hh < 16; ++hh) {
            int K = ((hh >> 3) << 4) + (g << 3) + (hh & 7);     // B 16-bit layout
            bf[hh] = (K < 9) ? ldsW[K*64 + (t*16 + (lane & 15))] : (_Float16)0.f;
        }
        v8f acc = {};
        acc = __builtin_amdgcn_wmma_f32_16x16x32_f16(
            false, a, false, bf, (short)0, acc, false, false);

        // per-lane partial reduction over this lane's 8 c-rows
        float s1p = 0.f, s2p = 0.f;
        #pragma unroll
        for (int j = 0; j < 8; ++j) {
            int c = wave*16 + j + (g << 3);     // C layout row = in-channel
            float v = acc[j];
            s1p += (c != 0) ? v : 0.f;                  // sum over c >= 1
            s2p += (c != 0) ? v*v : -v*v;               // Minkowski <avg,avg>
        }
        // fold the two lane-halves (same o set, complementary c halves)
        s1p += __shfl_xor(s1p, 16, 32);
        s2p += __shfl_xor(s2p, 16, 32);
        if ((lane & 16) == 0) {                 // one atomic per (tile, o)
            int o = t*16 + lane;
            atomicAdd(&s1[o], s1p);
            atomicAdd(&s2[o], s2p);
        }
    }
    __syncthreads();

    if (tid < 64) {
        float denom = __builtin_amdgcn_sqrtf(fmaxf(fabsf(s2[tid]), 1e-8f));
        mvec[tid] = s1[tid] / (denom * 63.0f);  // mean of cent[...,1:]
    }
    __syncthreads();
    if (tid == 0) {
        float acc = 0.f;
        for (int o = 1; o < 64; ++o) acc += mvec[o]*mvec[o];
        msum = acc;
    }
    __syncthreads();
    if (tid < 64)
        out[p*64 + tid] = (tid == 0) ? __builtin_amdgcn_sqrtf(1.0f + msum)
                                     : mvec[tid];
}

extern "C" void kernel_launch(void* const* d_in, const int* in_sizes, int n_in,
                              void* d_out, int out_size, void* d_ws, size_t ws_size,
                              hipStream_t stream) {
    const float* x    = (const float*)d_in[0];   // (8,56,56,64) f32
    const float* kern = (const float*)d_in[1];   // (64,64) f32
    float*       out  = (float*)d_out;           // (8,56,56,64) f32

    _Float16* xpad   = (_Float16*)d_ws;
    _Float16* dpad   = xpad + XPAD_ELEMS;
    _Float16* kern16 = dpad + XPAD_ELEMS;        // 4096 f16

    const int n16 = (int)((2u * XPAD_ELEMS * sizeof(_Float16)) / 16u);
    zero_ws_kernel<<<512, 256, 0, stream>>>((uint4*)d_ws, n16);
    cvt_kern_kernel<<<16, 256, 0, stream>>>(kern, kern16);
    lorentz_dists_kernel<<<NPIX/16, 128, 0, stream>>>(x, kern16, xpad, dpad);
    lorentz_agg_kernel<<<NPIX, 128, 0, stream>>>(xpad, dpad, out);
}
